// EncoderRNN_87763361726896
// MI455X (gfx1250) — matrix operand order repaired and checked by reading, not compiled, not run on hardware
//
#include <hip/hip_runtime.h>
#include <hip/hip_bf16.h>
#include <math.h>

typedef _Float16 v16h __attribute__((ext_vector_type(16)));
typedef float    v8f  __attribute__((ext_vector_type(8)));

#define T_LEN  8192
#define D_IN   256
#define HID    128
#define G4     512
#define KLAT   20
#define MITERS 10

#define USE_ASYNC_LDS 1

__device__ __forceinline__ v8f wmma_f16(v16h a, v16h b, v8f c) {
  // D = A(16x32 f16) * B(32x16 f16) + C(16x16 f32)
  return __builtin_amdgcn_wmma_f32_16x16x32_f16(false, a, false, b, (short)0, c, false, false);
}

__device__ __forceinline__ float u01hash(unsigned a, unsigned b) {
  unsigned x = a * 0x9E3779B9u + b * 0x85EBCA77u + 0x7F4A7C15u;
  x ^= x >> 16; x *= 0x85EBCA6Bu; x ^= x >> 13; x *= 0xC2B2AE35u; x ^= x >> 16;
  return ((x >> 8) + 1u) * (1.0f / 16777218.0f);   // (0,1)
}
__device__ __forceinline__ float gumbelf(unsigned a, unsigned b) {
  float u = u01hash(a, b);
  return -__logf(-__logf(u));
}

// ---------------------------------------------------------------- f32 -> f16
__global__ void cvt_f32_f16(const float* __restrict__ src, _Float16* __restrict__ dst, int n) {
  for (int i = blockIdx.x * blockDim.x + threadIdx.x; i < n; i += gridDim.x * blockDim.x)
    dst[i] = (_Float16)src[i];
}

// ------------------------------------------------- generic WMMA GEMM + bias
// C(MxN,f32) = A(MxK,f16,row-major) @ W(NxK,f16,row-major)^T + bias, opt relu.
// block = 128 thr (4 waves); each wave one 16x16 tile; grid = (M/16, N/64).
__global__ __launch_bounds__(128)
void wmma_gemm_bias(const _Float16* __restrict__ A, const _Float16* __restrict__ W,
                    const float* __restrict__ bias, float* __restrict__ C,
                    int M, int N, int K, int relu)
{
  __shared__ __align__(16) _Float16 As[16 * 32];
  const int tid   = threadIdx.x;
  const int lane  = tid & 31;
  const int wave  = tid >> 5;
  const int mbase = blockIdx.x * 16;
  const int nbase = blockIdx.y * 64 + wave * 16;
  const int r     = lane & 15;
  const int kb8   = (lane >> 4) * 8;    // unused; A-frag uses kb16 (k = 16h + s)
  const int nB    = lane & 15;
  const int kb16  = (lane >> 4) * 16;   // A/B-frag contiguous K split (k = 16h + s)

  v8f acc = {};
  for (int k0 = 0; k0 < K; k0 += 32) {
#if USE_ASYNC_LDS
    // stage the shared 16x32 A tile with CDNA5 async global->LDS copies:
    // 64 lanes x 16B. ASYNCcnt tracks completion.
    if (tid < 64) {
      int base = tid * 8;                     // half index (16B per lane)
      int row = base >> 5, col = base & 31;
      unsigned long long gaddr =
          (unsigned long long)(size_t)(A + (size_t)(mbase + row) * K + k0 + col);
      unsigned laddr = (unsigned)(size_t)(&As[base]);   // LDS byte offset
      asm volatile("global_load_async_to_lds_b128 %0, %1, off"
                   :: "v"(laddr), "v"(gaddr) : "memory");
    }
    asm volatile("s_wait_asynccnt 0" ::: "memory");
    __syncthreads();
#else
    { // cooperatively stage the shared 16x32 A tile in LDS
      int base = tid * 4;
      int row = base >> 5, col = base & 31;
      const _Float16* s = A + (size_t)(mbase + row) * K + k0 + col;
      As[base + 0] = s[0]; As[base + 1] = s[1]; As[base + 2] = s[2]; As[base + 3] = s[3];
    }
    __syncthreads();
#endif
    v16h af, bf;
#pragma unroll
    for (int i = 0; i < 8; ++i) {
      int kk = kb16 + 2 * i;
      af[2 * i]     = As[r * 32 + kk];
      af[2 * i + 1] = As[r * 32 + kk + 1];
    }
    const _Float16* wp = W + (size_t)(nbase + nB) * K + k0 + kb16;
#pragma unroll
    for (int i = 0; i < 8; ++i) {
      bf[2 * i]     = wp[2 * i];
      bf[2 * i + 1] = wp[2 * i + 1];
    }
    acc = wmma_f16(af, bf, acc);
    __syncthreads();
  }
  const int col = nbase + nB;
  const int rb  = (lane >> 4) * 8;
  float bv = bias ? bias[col] : 0.f;
#pragma unroll
  for (int vi = 0; vi < 8; ++vi) {
    float v = acc[vi] + bv;
    if (relu) v = fmaxf(v, 0.f);
    C[(size_t)(mbase + rb + vi) * N + col] = v;
  }
}

// ---------------------------------------------- row layernorm (128) -> f16
__global__ __launch_bounds__(256)
void layernorm_to_h(const float* __restrict__ X, const float* __restrict__ g,
                    const float* __restrict__ b, _Float16* __restrict__ Y)
{
  const int tid = threadIdx.x, lane = tid & 31, wv = tid >> 5;
  const int row = blockIdx.x * 8 + wv;
  const float* xr = X + (size_t)row * HID;
  float v[4]; float s = 0.f;
#pragma unroll
  for (int j = 0; j < 4; ++j) { v[j] = xr[lane + 32 * j]; s += v[j]; }
#pragma unroll
  for (int m = 16; m >= 1; m >>= 1) s += __shfl_xor(s, m, 32);
  float mu = s * (1.f / HID);
  float q = 0.f;
#pragma unroll
  for (int j = 0; j < 4; ++j) { float d = v[j] - mu; q += d * d; }
#pragma unroll
  for (int m = 16; m >= 1; m >>= 1) q += __shfl_xor(q, m, 32);
  float rinv = rsqrtf(q * (1.f / HID) + 1e-5f);
  _Float16* yr = Y + (size_t)row * HID;
#pragma unroll
  for (int j = 0; j < 4; ++j) {
    int c = lane + 32 * j;
    yr[c] = (_Float16)((v[j] - mu) * rinv * g[c] + b[c]);
  }
}

// ------------------------------------- dual LSTM scan (rnn_mask & seg_mask)
// One block, 512 thr = 16 waves. W_hh held as WMMA A-frags in VGPRs; per step
// gates(512x2) = W_hh(512x128) x [h1 h2] via 8 WMMAs/wave (cols 0/1 of B).
// pre / bias / mask loads are hoisted ahead of the WMMAs to hide latency.
__global__ __launch_bounds__(512, 1)
void lstm_scan2(const float* __restrict__ pre, const float* __restrict__ Whh,
                const float* __restrict__ bhh,
                const float* __restrict__ rnn_mask, const float* __restrict__ seg_mask,
                _Float16* __restrict__ outs_h, float* __restrict__ hfin, int T)
{
  __shared__ __align__(16) _Float16 hlds[2][HID];   // masked h, f16 (B operand)
  __shared__ float glds[2][G4];                     // gate pre-activations

  const int tid  = threadIdx.x;
  const int lane = tid & 31;
  const int wave = tid >> 5;                 // 0..15, owns gates 32w..32w+31
  const int r    = lane & 15;
  const int kb8  = (lane >> 4) * 8;
  const int nB   = lane & 15;                // B column (0/1 = state)
  const int kb16 = (lane >> 4) * 16;
  const int rb   = (lane >> 4) * 8;

  // resident A fragments: 2 row-tiles x 4 K-tiles of W_hh (f32 -> f16 once)
  v16h afrag[2][4];
#pragma unroll
  for (int tt = 0; tt < 2; ++tt)
#pragma unroll
    for (int kt = 0; kt < 4; ++kt) {
      const float* wr = Whh + (size_t)(32 * wave + 16 * tt + r) * HID + 32 * kt;
#pragma unroll
      for (int i = 0; i < 8; ++i) {
        int kk = kb16 + 2 * i;
        afrag[tt][kt][2 * i]     = (_Float16)wr[kk];
        afrag[tt][kt][2 * i + 1] = (_Float16)wr[kk + 1];
      }
    }

  const int s = tid >> 7;                    // state (threads 0..255)
  const int e = tid & 127;                   // hidden element
  float cc = 0.f;
  float bb0 = 0.f, bb1 = 0.f, bb2v = 0.f, bb3 = 0.f;
  if (tid < 256) {
    hlds[s][e] = (_Float16)0.f;
    bb0 = bhh[e]; bb1 = bhh[HID + e]; bb2v = bhh[2 * HID + e]; bb3 = bhh[3 * HID + e];
  }
  __syncthreads();

  for (int t = 0; t < T; ++t) {
    // issue pre/mask loads early; they complete while the WMMAs run
    float p0v = 0.f, p1v = 0.f, p2v = 0.f, p3v = 0.f, mval = 0.f;
    if (tid < 256) {
      const float* pt = pre + (size_t)t * G4;
      if (t + 1 < T) __builtin_prefetch(pt + G4, 0, 0);
      p0v = pt[e]; p1v = pt[HID + e]; p2v = pt[2 * HID + e]; p3v = pt[3 * HID + e];
      mval = (s == 0) ? rnn_mask[t] : seg_mask[t];
    }

    v8f acc0 = {}, acc1 = {};
#pragma unroll
    for (int kt = 0; kt < 4; ++kt) {
      v16h bf = {};
      if (nB < 2) {
        const _Float16* hl = &hlds[nB][32 * kt + kb16];
#pragma unroll
        for (int i = 0; i < 16; ++i) bf[i] = hl[i];
      }
      acc0 = wmma_f16(afrag[0][kt], bf, acc0);
      acc1 = wmma_f16(afrag[1][kt], bf, acc1);
    }
    if (nB < 2) {                            // cols 0/1 hold the two states
#pragma unroll
      for (int vi = 0; vi < 8; ++vi) {
        glds[nB][32 * wave +      rb + vi] = acc0[vi];
        glds[nB][32 * wave + 16 + rb + vi] = acc1[vi];
      }
    }
    __syncthreads();
    if (tid < 256) {
      float gi = glds[s][e]           + p0v + bb0;
      float gf = glds[s][HID + e]     + p1v + bb1;
      float gg = glds[s][2 * HID + e] + p2v + bb2v;
      float go = glds[s][3 * HID + e] + p3v + bb3;
      float sf = 1.f / (1.f + __expf(-gf));
      float si = 1.f / (1.f + __expf(-gi));
      float so = 1.f / (1.f + __expf(-go));
      cc = sf * cc + si * tanhf(gg);
      float ho = so * tanhf(cc);
      if (s == 0) outs_h[(size_t)t * HID + e] = (_Float16)ho;   // pre-mask out
      hlds[s][e] = (_Float16)(ho * mval);                        // masked carry
    }
    __syncthreads();
  }
  if (tid >= 128 && tid < 256) hfin[e] = (float)hlds[1][e];     // final seg h
}

// ------------------------------- hb[t] = relu(outs@Wb1+b) @ wb2 + b2 + gumbel
__global__ __launch_bounds__(256)
void hb_rowdot(const float* __restrict__ hb1, const float* __restrict__ wb2,
               const float* __restrict__ bb2, float* __restrict__ hb, int m)
{
  const int tid = threadIdx.x, lane = tid & 31, wv = tid >> 5;
  const int row = blockIdx.x * 8 + wv;
  const float* hr = hb1 + (size_t)row * 64;
  float v = hr[lane] * wb2[lane] + hr[lane + 32] * wb2[lane + 32];
#pragma unroll
  for (int mm = 16; mm >= 1; mm >>= 1) v += __shfl_xor(v, mm, 32);
  if (lane == 0)
    hb[row] = v + bb2[0] + gumbelf(0xB0B0u + 2u * m + 1u, (unsigned)row);
}

// ------------------------------------------- latent head + gumbel softmax(K)
__global__ __launch_bounds__(64)
void head_z(const float* __restrict__ h, const float* __restrict__ Wz1,
            const float* __restrict__ bz1, const float* __restrict__ Wz2,
            const float* __restrict__ bz2, float* __restrict__ lat, int m)
{
  __shared__ float h1[64];
  __shared__ float hz[KLAT];
  const int tid = threadIdx.x;
  float a = bz1[tid];
  for (int k = 0; k < HID; ++k) a += h[k] * Wz1[tid * HID + k];
  h1[tid] = fmaxf(a, 0.f);
  __syncthreads();
  if (tid < KLAT) {
    float b = bz2[tid];
    for (int j = 0; j < 64; ++j) b += h1[j] * Wz2[tid * 64 + j];
    hz[tid] = b + gumbelf(0xA5A5u + 2u * m, (unsigned)tid);
  }
  __syncthreads();
  if (tid == 0) {
    float mx = hz[0];
    for (int k = 1; k < KLAT; ++k) mx = fmaxf(mx, hz[k]);
    float sm = 0.f;
    for (int k = 0; k < KLAT; ++k) sm += __expf(hz[k] - mx);
    float inv = 1.f / sm;
    for (int k = 0; k < KLAT; ++k) lat[m * KLAT + k] = __expf(hz[k] - mx) * inv;
  }
}

// ---------------- softmax(T) + cumsum + mask update, writes probs/segs rows
__global__ __launch_bounds__(1024, 1)
void update_masks(float* __restrict__ hb, float* __restrict__ running,
                  float* __restrict__ rnn_mask, float* __restrict__ seg_mask,
                  float* __restrict__ probs, float* __restrict__ segs, int m)
{
  __shared__ float red[1024];
  const int tid = threadIdx.x;
  const int base = tid * 8;

  float lm = -3.4e38f;
  float xv[8];
#pragma unroll
  for (int j = 0; j < 8; ++j) { xv[j] = hb[base + j]; lm = fmaxf(lm, xv[j]); }
  red[tid] = lm; __syncthreads();
  for (int s2 = 512; s2 >= 1; s2 >>= 1) {
    if (tid < s2) red[tid] = fmaxf(red[tid], red[tid + s2]);
    __syncthreads();
  }
  float mx = red[0]; __syncthreads();

  float ls = 0.f; float ev[8];
#pragma unroll
  for (int j = 0; j < 8; ++j) { ev[j] = __expf(xv[j] - mx); ls += ev[j]; }
  red[tid] = ls; __syncthreads();
  for (int s2 = 512; s2 >= 1; s2 >>= 1) {
    if (tid < s2) red[tid] += red[tid + s2];
    __syncthreads();
  }
  float inv = 1.f / red[0]; __syncthreads();

  float p = 0.f; float cs[8];
#pragma unroll
  for (int j = 0; j < 8; ++j) { p += ev[j] * inv; cs[j] = p; }
  red[tid] = p; __syncthreads();
  for (int off = 1; off < 1024; off <<= 1) {          // Hillis-Steele scan
    float a = (tid >= off) ? red[tid - off] : 0.f;
    __syncthreads();
    red[tid] += a;
    __syncthreads();
  }
  float carry = (tid > 0) ? red[tid - 1] : 0.f;

#pragma unroll
  for (int j = 0; j < 8; ++j) {
    int t = base + j;
    float bv  = ev[j] * inv;
    float csj = carry + cs[j];
    float run = running[t];
    probs[(size_t)(m + 1) * T_LEN + t] = bv;
    float seg = run * (1.f - csj);
    segs[(size_t)(m + 1) * T_LEN + t] = seg;
    seg_mask[t] = seg;
    rnn_mask[t] = run;
    running[t]  = run * csj;
  }
}

// ------------------------------------------------------------ init state/out
__global__ void init_state(float* __restrict__ running, float* __restrict__ rnn_mask,
                           float* __restrict__ seg_mask, float* __restrict__ probs,
                           float* __restrict__ segs)
{
  for (int t = blockIdx.x * blockDim.x + threadIdx.x; t < T_LEN;
       t += gridDim.x * blockDim.x) {
    running[t] = 1.f; rnn_mask[t] = 1.f; seg_mask[t] = 1.f;
    probs[t] = (t == 0) ? 1.f : 0.f;     // row 0 = p0
    segs[t]  = 1.f;                      // row 0 = ones
  }
}

// ============================================================== host driver
extern "C" void kernel_launch(void* const* d_in, const int* in_sizes, int n_in,
                              void* d_out, int out_size, void* d_ws, size_t ws_size,
                              hipStream_t stream) {
  const float* x     = (const float*)d_in[0];
  const float* W_pre = (const float*)d_in[1];
  const float* b_pre = (const float*)d_in[2];
  const float* ln_g  = (const float*)d_in[3];
  const float* ln_b  = (const float*)d_in[4];
  const float* W_ih  = (const float*)d_in[5];
  const float* W_hh  = (const float*)d_in[6];
  const float* b_ih  = (const float*)d_in[7];
  const float* b_hh  = (const float*)d_in[8];
  const float* W_z1  = (const float*)d_in[9];
  const float* b_z1  = (const float*)d_in[10];
  const float* W_z2  = (const float*)d_in[11];
  const float* b_z2  = (const float*)d_in[12];
  const float* W_b1  = (const float*)d_in[13];
  const float* b_b1  = (const float*)d_in[14];
  const float* W_b2  = (const float*)d_in[15];
  const float* b_b2  = (const float*)d_in[16];

  char* wp = (char*)d_ws;
  auto carve = [&](size_t bytes) { void* p = (void*)wp; wp += (bytes + 255) & ~(size_t)255; return p; };
  _Float16* xh      = (_Float16*)carve((size_t)T_LEN * D_IN * 2);
  _Float16* wpreh   = (_Float16*)carve((size_t)HID * D_IN * 2);
  float*    inp_pre = (float*)   carve((size_t)T_LEN * HID * 4);
  _Float16* inp_h   = (_Float16*)carve((size_t)T_LEN * HID * 2);
  _Float16* wih_h   = (_Float16*)carve((size_t)G4 * HID * 2);
  _Float16* wb1_h   = (_Float16*)carve((size_t)64 * HID * 2);
  float*    pre     = (float*)   carve((size_t)T_LEN * G4 * 4);
  _Float16* outs_h  = (_Float16*)carve((size_t)T_LEN * HID * 2);
  float*    hb1     = (float*)   carve((size_t)T_LEN * 64 * 4);
  float*    hb      = (float*)   carve((size_t)T_LEN * 4);
  float*    hfin    = (float*)   carve((size_t)HID * 4);
  float*    running = (float*)   carve((size_t)T_LEN * 4);
  float*    rnn_m   = (float*)   carve((size_t)T_LEN * 4);
  float*    seg_m   = (float*)   carve((size_t)T_LEN * 4);

  float* out   = (float*)d_out;
  float* probs = out;                                   // (11, 8192)
  float* lat   = out + (size_t)(MITERS + 1) * T_LEN;    // (10, 20)
  float* segs  = lat + (size_t)MITERS * KLAT;           // (11, 8192)

  // f32 -> f16 weight/activation conversions
  cvt_f32_f16<<<256, 256, 0, stream>>>(x,     xh,    T_LEN * D_IN);
  cvt_f32_f16<<<64,  256, 0, stream>>>(W_pre, wpreh, HID * D_IN);
  cvt_f32_f16<<<64,  256, 0, stream>>>(W_ih,  wih_h, G4 * HID);
  cvt_f32_f16<<<32,  256, 0, stream>>>(W_b1,  wb1_h, 64 * HID);

  // inp_pre = x @ W_pre.T + b_pre   (8192x256 -> 8192x128)
  wmma_gemm_bias<<<dim3(T_LEN / 16, HID / 64), 128, 0, stream>>>(
      xh, wpreh, b_pre, inp_pre, T_LEN, HID, D_IN, 0);
  // inp = LN(inp_pre) -> f16
  layernorm_to_h<<<T_LEN / 8, 256, 0, stream>>>(inp_pre, ln_g, ln_b, inp_h);
  // pre = inp @ W_ih.T + b_ih       (8192x128 -> 8192x512)
  wmma_gemm_bias<<<dim3(T_LEN / 16, G4 / 64), 128, 0, stream>>>(
      inp_h, wih_h, b_ih, pre, T_LEN, G4, HID, 0);

  init_state<<<32, 256, 0, stream>>>(running, rnn_m, seg_m, probs, segs);

  for (int m = 0; m < MITERS; ++m) {
    lstm_scan2<<<1, 512, 0, stream>>>(pre, W_hh, b_hh, rnn_m, seg_m, outs_h, hfin, T_LEN);
    // hb1 = relu(outs @ W_b1.T + b_b1)  (8192x128 -> 8192x64)
    wmma_gemm_bias<<<dim3(T_LEN / 16, 1), 128, 0, stream>>>(
        outs_h, wb1_h, b_b1, hb1, T_LEN, 64, HID, 1);
    hb_rowdot<<<T_LEN / 8, 256, 0, stream>>>(hb1, W_b2, b_b2, hb, m);
    head_z<<<1, 64, 0, stream>>>(hfin, W_z1, b_z1, W_z2, b_z2, lat, m);
    update_masks<<<1, 1024, 0, stream>>>(hb, running, rnn_m, seg_m, probs, segs, m);
  }
}
